// ContrastiveCorrelationLoss_31696858645016
// MI455X (gfx1250) — matrix operand order repaired and checked by reading, not compiled, not run on hardware
//
#include <hip/hip_runtime.h>
#include <hip/hip_bf16.h>

// ---------------------------------------------------------------------------
// ContrastiveCorrelationLoss, fused WMMA version for gfx1250 (MI455X)
// ---------------------------------------------------------------------------
#define BB    16
#define SS    32
#define NPOS  1024      // S*S sampled positions per image
#define CFD   768       // feature channels
#define CCD   512       // code channels
#define HH    28
#define WW    28

typedef __attribute__((ext_vector_type(16))) __bf16 v16bf;
typedef __attribute__((ext_vector_type(8)))  float  v8f;

static constexpr size_t F_ELEMS = (size_t)BB * NPOS * CFD;   // bf16 elems per F-matrix
static constexpr size_t C_ELEMS = (size_t)BB * NPOS * CCD;   // bf16 elems per C-matrix
// ws layout: [F1|Fpos|Fneg0|Fneg1][C1|Cpos|Cneg0|Cneg1][16 float accumulators]
static constexpr size_t ACC_OFF_BYTES = (4 * F_ELEMS + 4 * C_ELEMS) * sizeof(__bf16);

__device__ inline v8f zero8() {
  v8f z = {0.f, 0.f, 0.f, 0.f, 0.f, 0.f, 0.f, 0.f};
  return z;
}

__device__ inline v8f wmma_bf16(v16bf a, v16bf b, v8f c) {
  return __builtin_amdgcn_wmma_f32_16x16x32_bf16(
      /*neg_a=*/false, a, /*neg_b=*/false, b,
      /*c_mod=*/(short)0, c, /*reuse_a=*/false, /*reuse_b=*/false);
}

// A fragment (16x32 bf16): lane m=lane&15 is the row; hs=lane>>4 selects
// K chunks [hs*8, hs*8+8) and [16+hs*8, 16+hs*8+8)  (ISA 7.12.2 layout)
__device__ inline v16bf load_a(const __bf16* mat, int ld, int row, int kb,
                               int m, int hs) {
  union { v16bf v; uint4 u[2]; } f;
  const __bf16* p = mat + (size_t)(row + m) * ld + kb + hs * 8;
  f.u[0] = *(const uint4*)p;
  f.u[1] = *(const uint4*)(p + 16);
  return f.v;
}

// B fragment (32x16 bf16): lane m is the column; hs selects K chunks
// [hs*8, hs*8+8) and [16+hs*8, 16+hs*8+8). Right matrix is row-major [col][k] (B = RMᵀ).
__device__ inline v16bf load_b(const __bf16* mat, int ld, int col, int kb,
                               int m, int hs) {
  union { v16bf v; uint4 u[2]; } f;
  const __bf16* p = mat + (size_t)(col + m) * ld + kb + hs * 8;
  f.u[0] = *(const uint4*)p;
  f.u[1] = *(const uint4*)(p + 16);
  return f.v;
}

// ---------------------------------------------------------------------------
// Kernel 1: bilinear sample + per-position L2 normalize -> bf16 matrices
// One block (256 thr) per (tensor, batch, position).
// tensors: 0 F@c1, 1 Fpos@c2, 2 Fneg0@c2, 3 Fneg1@c2, 4..7 same for code
// ---------------------------------------------------------------------------
__global__ __launch_bounds__(256)
void sample_norm_kernel(const float* __restrict__ feats,
                        const float* __restrict__ feats_pos,
                        const float* __restrict__ code,
                        const float* __restrict__ code_pos,
                        const float* __restrict__ coords1,
                        const float* __restrict__ coords2,
                        const int*   __restrict__ perms,
                        __bf16* __restrict__ ws_bf) {
  const int bid = blockIdx.x;
  const int tensor = bid >> 14;        // 0..7
  const int n  = (bid >> 10) & (BB - 1);
  const int p  = bid & (NPOS - 1);
  const int tid = threadIdx.x;

  const bool isF = tensor < 4;
  const int  C   = isF ? CFD : CCD;
  const int  which = tensor & 3;       // 0:self/c1  1:pos/c2  2,3:neg/c2
  const float* src = isF ? ((which == 1) ? feats_pos : feats)
                         : ((which == 1) ? code_pos  : code);
  const float* crd = (which == 0) ? coords1 : coords2;
  int nsrc = n;
  if (which >= 2) nsrc = perms[(which - 2) * BB + n];

  const size_t ci = (((size_t)n * SS + (p >> 5)) * SS + (p & 31)) * 2;
  const float x = crd[ci + 0] * (float)(WW - 1);   // (c*2-1 +1)*0.5*(W-1)
  const float y = crd[ci + 1] * (float)(HH - 1);
  const float x0f = floorf(x), y0f = floorf(y);
  const float wx = x - x0f, wy = y - y0f;
  const int x0 = min(max((int)x0f,     0), WW - 1);
  const int x1 = min(max((int)x0f + 1, 0), WW - 1);
  const int y0 = min(max((int)y0f,     0), HH - 1);
  const int y1 = min(max((int)y0f + 1, 0), HH - 1);
  const float w00 = (1.f - wx) * (1.f - wy);
  const float w01 = (1.f - wx) * wy;
  const float w10 = wx * (1.f - wy);
  const float w11 = wx * wy;

  const float* base = src + (size_t)nsrc * C * (HH * WW);
  float vv[3];
  float ssq = 0.f;
  int cnt = 0;
  for (int c = tid; c < C; c += 256) {
    const float* ip = base + (size_t)c * (HH * WW);
    float v = w00 * ip[y0 * WW + x0] + w01 * ip[y1 * WW + x0] +
              w10 * ip[y0 * WW + x1] + w11 * ip[y1 * WW + x1];
    vv[cnt++] = v;
    ssq += v * v;
  }

  __shared__ float warp_s[8];
  __shared__ float inv_s;
  for (int mk = 1; mk < 32; mk <<= 1) ssq += __shfl_xor(ssq, mk, 32);
  if ((tid & 31) == 0) warp_s[tid >> 5] = ssq;
  __syncthreads();
  if (tid == 0) {
    float t = 0.f;
    for (int i = 0; i < 8; ++i) t += warp_s[i];
    inv_s = 1.f / fmaxf(sqrtf(t), 1e-10f);
  }
  __syncthreads();
  const float inv = inv_s;

  __bf16* dst = isF
      ? ws_bf + (size_t)which * F_ELEMS + ((size_t)n * NPOS + p) * CFD
      : ws_bf + 4 * F_ELEMS + (size_t)which * C_ELEMS + ((size_t)n * NPOS + p) * CCD;
  cnt = 0;
  for (int c = tid; c < C; c += 256)
    dst[c] = (__bf16)(vv[cnt++] * inv);
}

// ---------------------------------------------------------------------------
// Kernel 2: fused correlation + loss reductions.
// grid = (16 rowblocks, 16 batches, 4 helpers), 256 threads = 8 waves
// arranged as a 2x4 grid: wave (wvr, wvc) computes a 32x32 patch at
// rows rb*64 + wvr*32, cols pass*128 + wvc*32.  A fragments are shared by
// the 4 waves of a row group, B fragments by the 2 waves of a column group
// (WGP$/L2 reuse), cutting L2 read traffic ~2.7x vs 1-D wave layout.
// Each workgroup owns 64 complete rows -> row sums stay local.
// Accumulates per helper: [Σ relu*fd, Σ_rows rowmean_fd*rowsum_relu,
//                          Σ relu, Σ fd]
// ---------------------------------------------------------------------------
__global__ __launch_bounds__(256)
__attribute__((amdgpu_waves_per_eu(4)))
void corr_wmma_kernel(const __bf16* __restrict__ ws_bf,
                      float* __restrict__ accum) {
  const int h    = blockIdx.z;               // helper 0..3
  const int n    = blockIdx.y;               // batch
  const int rb   = blockIdx.x;               // row block (64 rows)
  const int tid  = threadIdx.x;
  const int wv   = tid >> 5;
  const int wvr  = wv >> 2;                  // wave row group 0..1
  const int wvc  = wv & 3;                   // wave col group 0..3
  const int lane = tid & 31;
  const int m    = lane & 15;
  const int hs   = lane >> 4;

  const __bf16* Fbase = ws_bf;
  const __bf16* Cbase = ws_bf + 4 * F_ELEMS;
  const int ridx = (h == 0) ? 0 : h;         // right-side matrix selector
  const __bf16* LF = Fbase + (size_t)n * NPOS * CFD;
  const __bf16* RF = Fbase + (size_t)ridx * F_ELEMS + (size_t)n * NPOS * CFD;
  const __bf16* LC = Cbase + (size_t)n * NPOS * CCD;
  const __bf16* RC = Cbase + (size_t)ridx * C_ELEMS + (size_t)n * NPOS * CCD;

  __shared__ float lds_fd[64];   // per-row Σ fd      (over all 1024 cols)
  __shared__ float lds_rl[64];   // per-row Σ relu(cd)
  __shared__ float lds_sc[3];    // Σ relu*fd, Σ relu, Σ fd
  if (tid < 64) { lds_fd[tid] = 0.f; lds_rl[tid] = 0.f; }
  if (tid < 3)  lds_sc[tid] = 0.f;
  __syncthreads();

  const int row0 = rb * 64 + wvr * 32;
  float sA = 0.f, sC = 0.f, sD = 0.f;

  for (int pass = 0; pass < 8; ++pass) {
    const int col0 = pass * 128 + wvc * 32;

    v8f fd[2][2], cd[2][2];
#pragma unroll
    for (int i = 0; i < 2; ++i)
#pragma unroll
      for (int j = 0; j < 2; ++j) { fd[i][j] = zero8(); cd[i][j] = zero8(); }

    // ---- feature correlation: K = 768
    for (int kb = 0; kb < CFD; kb += 32) {
      v16bf a0 = load_a(LF, CFD, row0,      kb, m, hs);
      v16bf a1 = load_a(LF, CFD, row0 + 16, kb, m, hs);
      v16bf b0 = load_b(RF, CFD, col0,      kb, m, hs);
      v16bf b1 = load_b(RF, CFD, col0 + 16, kb, m, hs);
      fd[0][0] = wmma_bf16(a0, b0, fd[0][0]);
      fd[0][1] = wmma_bf16(a0, b1, fd[0][1]);
      fd[1][0] = wmma_bf16(a1, b0, fd[1][0]);
      fd[1][1] = wmma_bf16(a1, b1, fd[1][1]);
    }
    // ---- code correlation: K = 512
    for (int kb = 0; kb < CCD; kb += 32) {
      v16bf a0 = load_a(LC, CCD, row0,      kb, m, hs);
      v16bf a1 = load_a(LC, CCD, row0 + 16, kb, m, hs);
      v16bf b0 = load_b(RC, CCD, col0,      kb, m, hs);
      v16bf b1 = load_b(RC, CCD, col0 + 16, kb, m, hs);
      cd[0][0] = wmma_bf16(a0, b0, cd[0][0]);
      cd[0][1] = wmma_bf16(a0, b1, cd[0][1]);
      cd[1][0] = wmma_bf16(a1, b0, cd[1][0]);
      cd[1][1] = wmma_bf16(a1, b1, cd[1][1]);
    }

    // ---- fused elementwise + row reductions
    // C/D layout: lane m is column; row within 16x16 tile = v + hs*8.
#pragma unroll
    for (int tr = 0; tr < 2; ++tr) {
#pragma unroll
      for (int v = 0; v < 8; ++v) {
        float f0 = fd[tr][0][v], f1 = fd[tr][1][v];
        float c0 = cd[tr][0][v], c1 = cd[tr][1][v];
        float r0 = fmaxf(c0, 0.f), r1 = fmaxf(c1, 0.f);
        sA += r0 * f0 + r1 * f1;
        sC += r0 + r1;
        sD += f0 + f1;
        float rs_f = f0 + f1;      // this lane's contribution to row sums
        float rs_r = r0 + r1;
#pragma unroll
        for (int mk = 1; mk < 16; mk <<= 1) {     // reduce over 16 columns
          rs_f += __shfl_xor(rs_f, mk, 32);
          rs_r += __shfl_xor(rs_r, mk, 32);
        }
        if (m == 0) {
          const int row_local = wvr * 32 + tr * 16 + hs * 8 + v;
          atomicAdd(&lds_fd[row_local], rs_f);
          atomicAdd(&lds_rl[row_local], rs_r);
        }
      }
    }
  }

  // wave-wide scalar reduction
  for (int mk = 1; mk < 32; mk <<= 1) {
    sA += __shfl_xor(sA, mk, 32);
    sC += __shfl_xor(sC, mk, 32);
    sD += __shfl_xor(sD, mk, 32);
  }
  if (lane == 0) {
    atomicAdd(&lds_sc[0], sA);
    atomicAdd(&lds_sc[1], sC);
    atomicAdd(&lds_sc[2], sD);
  }
  __syncthreads();

  if (tid == 0) {
    float bterm = 0.f;
    for (int r = 0; r < 64; ++r) bterm += lds_fd[r] * lds_rl[r];
    bterm *= (1.0f / (float)NPOS);               // rowmean_fd * rowsum_relu
    float* acc = accum + h * 4;
    atomicAdd(&acc[0], lds_sc[0]);               // Σ relu*fd
    atomicAdd(&acc[1], bterm);                   // Σ rowmean_fd*rowsum_relu
    atomicAdd(&acc[2], lds_sc[1]);               // Σ relu
    atomicAdd(&acc[3], lds_sc[2]);               // Σ fd
  }
}

// ---------------------------------------------------------------------------
__global__ void init_accum_kernel(float* accum) {
  if (threadIdx.x < 16) accum[threadIdx.x] = 0.f;
}

__global__ void finalize_kernel(const float* __restrict__ accum,
                                float* __restrict__ out) {
  if (threadIdx.x == 0) {
    const float shifts[4] = {0.18f, 0.12f, 0.46f, 0.46f};
    const float N = (float)BB * (float)NPOS * (float)NPOS;  // 16,777,216
    float loss[4];
    for (int h = 0; h < 4; ++h) {
      float A  = accum[h * 4 + 0];
      float Bt = accum[h * 4 + 1];
      float Cr = accum[h * 4 + 2];
      float Ds = accum[h * 4 + 3];
      float gmean = Ds / N;
      loss[h] = -(A - Bt + (gmean - shifts[h]) * Cr) / N;
    }
    out[0] = loss[0];
    out[1] = loss[1];
    out[2] = 0.5f * (loss[2] + loss[3]);
  }
}

// ---------------------------------------------------------------------------
extern "C" void kernel_launch(void* const* d_in, const int* in_sizes, int n_in,
                              void* d_out, int out_size, void* d_ws,
                              size_t ws_size, hipStream_t stream) {
  const float* orig_feats     = (const float*)d_in[0];
  const float* orig_feats_pos = (const float*)d_in[1];
  const float* orig_code      = (const float*)d_in[4];
  const float* orig_code_pos  = (const float*)d_in[5];
  const float* coords1        = (const float*)d_in[8];
  const float* coords2        = (const float*)d_in[9];
  const int*   perms          = (const int*)d_in[10];

  __bf16* ws_bf = (__bf16*)d_ws;
  float*  accum = (float*)((char*)d_ws + ACC_OFF_BYTES);
  float*  out   = (float*)d_out;

  init_accum_kernel<<<1, 32, 0, stream>>>(accum);
  sample_norm_kernel<<<dim3(8 * BB * NPOS), dim3(256), 0, stream>>>(
      orig_feats, orig_feats_pos, orig_code, orig_code_pos,
      coords1, coords2, perms, ws_bf);
  corr_wmma_kernel<<<dim3(16, BB, 4), dim3(256), 0, stream>>>(ws_bf, accum);
  finalize_kernel<<<1, 32, 0, stream>>>(accum, out);
}
